// AttentionDiff_90323162235124
// MI455X (gfx1250) — matrix operand order repaired and checked by reading, not compiled, not run on hardware
//
#include <hip/hip_runtime.h>
#include <math.h>

// Problem constants (from reference): B=4, T=4096, C=1024, HEAD_SIZE=64
#define T_SEQ 4096
#define C_DIM 1024
#define HS    64
#define D2    128   // 2*HEAD_SIZE

typedef __attribute__((ext_vector_type(16))) __bf16 v16bf;
typedef __attribute__((ext_vector_type(8)))  float  v8f;
typedef __attribute__((ext_vector_type(4)))  int    v4i;

#if defined(__has_builtin)
#  if __has_builtin(__builtin_amdgcn_global_load_async_to_lds_b128)
#    define HAVE_ASYNC_LDS 1
#  endif
#  if __has_builtin(__builtin_amdgcn_s_wait_asynccnt)
#    define HAVE_WAIT_ASYNC 1
#  endif
#endif

typedef __attribute__((address_space(1))) v4i as1_v4i;
typedef __attribute__((address_space(3))) v4i as3_v4i;

// 16-byte global -> LDS copy. Async (ASYNCcnt-tracked, VGPR-bypassing) when
// the gfx1250 builtin is available, else synchronous fallback.
__device__ __forceinline__ void async_copy16(const unsigned short* g, unsigned short* l) {
#if defined(HAVE_ASYNC_LDS)
  __builtin_amdgcn_global_load_async_to_lds_b128((as1_v4i*)g, (as3_v4i*)l, 0, 0);
#else
  *(uint4*)l = *(const uint4*)g;
#endif
}

__device__ __forceinline__ void wait_async_lds() {
#if defined(HAVE_ASYNC_LDS)
#  if defined(HAVE_WAIT_ASYNC)
  __builtin_amdgcn_s_wait_asynccnt(0);
#  else
  asm volatile("s_wait_asynccnt 0x0" ::: "memory");
#  endif
#endif
}

__device__ __forceinline__ unsigned short f2bf(float f) {
  unsigned u = __float_as_uint(f);
  u += 0x7FFFu + ((u >> 16) & 1u);            // round-to-nearest-even
  return (unsigned short)(u >> 16);
}

// A-matrix fragment (16Mx32K bf16), per ISA 7.12.2:
// lanes 0-15: M=lane, VGPR g in 0..3 -> K=2g,2g+1 ; g in 4..7 -> K=16+2(g-4),...
// lanes 16-31: same M, K offset +8.
__device__ __forceinline__ v16bf load_a(const unsigned short* base, int row,
                                        int ld, int kbase, int lane) {
  int off = (lane & 16) ? 8 : 0;
  union { v16bf v; unsigned u[8]; } r;
  const unsigned short* p = base + (size_t)row * ld + kbase + off;
#pragma unroll
  for (int g = 0; g < 8; ++g) {
    int kk = (g & 3) * 2 + (g >> 2) * 16;
    r.u[g] = *(const unsigned*)(p + kk);
  }
  return r.v;
}

// B-matrix fragment (32Kx16N bf16): lane n holds column n; lanes 0-15
// VGPR g in 0..3 -> K=2g,2g+1 ; g in 4..7 -> K=16+2(g-4),...; lanes 16-31 K offset +8.
// Source stored [col][K] row-major.
__device__ __forceinline__ v16bf load_b(const unsigned short* base, int col,
                                        int ld, int kbase, int lane) {
  int off = (lane & 16) ? 8 : 0;
  union { v16bf v; unsigned u[8]; } r;
  const unsigned short* p = base + (size_t)col * ld + kbase + off;
#pragma unroll
  for (int g = 0; g < 8; ++g) r.u[g] = *(const unsigned*)(p + (g & 3) * 2 + (g >> 2) * 16);
  return r.v;
}

__device__ __forceinline__ v8f wmma_bf16(v16bf a, v16bf b, v8f c) {
  return __builtin_amdgcn_wmma_f32_16x16x32_bf16(false, a, false, b,
                                                 (short)0, c, false, false);
}

// ---------------------------------------------------------------------------
// Kernel 1: qkv projection. grid=(M/128, 3). blockIdx.y selects Wq/Wk/Wv.
// Outputs bf16: qb,kb row-major (B*T,128); v transposed (B,128,T) for PV WMMA.
// ---------------------------------------------------------------------------
__global__ __launch_bounds__(256) void qkv_proj_kernel(
    const float* __restrict__ x,  const float* __restrict__ Wq,
    const float* __restrict__ Wk, const float* __restrict__ Wv,
    unsigned short* __restrict__ qb, unsigned short* __restrict__ kb,
    unsigned short* __restrict__ vbt) {
  __shared__ __align__(16) unsigned short xs[128 * 32];   // [row][k]
  __shared__ __align__(16) unsigned short wst[128 * 32];  // [col][k] (transposed)

  const int tid = threadIdx.x;
  const int wave = tid >> 5, lane = tid & 31, laneN = lane & 15;
  const int half8 = (lane & 16) ? 8 : 0;
  const int rowbase = blockIdx.x * 128;
  const int wsel = blockIdx.y;
  const float* W = (wsel == 0) ? Wq : ((wsel == 1) ? Wk : Wv);

  v8f vzero = {};
  v8f acc[8];
#pragma unroll
  for (int o = 0; o < 8; ++o) acc[o] = vzero;

  for (int kt = 0; kt < C_DIM / 32; ++kt) {
    {
      // stage x tile (128x32 f32 -> bf16)
      int r = tid >> 1, c0 = (tid & 1) * 16;
      const float* src = x + (size_t)(rowbase + r) * C_DIM + kt * 32 + c0;
#pragma unroll
      for (int i = 0; i < 16; ++i) xs[r * 32 + c0 + i] = f2bf(src[i]);
      // stage W tile (32x128) transposed -> wst[col][k]
      int kr = tid >> 3, wc0 = (tid & 7) * 16;
      const float* wsrc = W + (size_t)(kt * 32 + kr) * D2 + wc0;
#pragma unroll
      for (int i = 0; i < 16; ++i) wst[(wc0 + i) * 32 + kr] = f2bf(wsrc[i]);
      if (kt + 1 < C_DIM / 32) {
        __builtin_prefetch(src + 32, 0, 1);          // next x K-step
        __builtin_prefetch(wsrc + 32 * D2, 0, 1);    // next W K-step
      }
    }
    __syncthreads();
    v16bf a = load_a(xs, wave * 16 + laneN, 32, 0, lane);
#pragma unroll
    for (int o = 0; o < 8; ++o) {
      v16bf bfr = load_b(wst, o * 16 + laneN, 32, 0, lane);
      acc[o] = wmma_bf16(a, bfr, acc[o]);
    }
    __syncthreads();
  }

  const int mrow = rowbase + wave * 16;
#pragma unroll
  for (int o = 0; o < 8; ++o) {
    int col = o * 16 + laneN;
#pragma unroll
    for (int r = 0; r < 8; ++r) {
      int row = mrow + r + half8;
      unsigned short val = f2bf(acc[o][r]);
      if (wsel == 0)      qb[(size_t)row * D2 + col] = val;
      else if (wsel == 1) kb[(size_t)row * D2 + col] = val;
      else {
        int bidx = row >> 12, t = row & (T_SEQ - 1);
        vbt[((size_t)bidx * D2 + col) * T_SEQ + t] = val;
      }
    }
  }
}

// ---------------------------------------------------------------------------
// Kernel 2: fused differential flash attention, double-buffered async K/V
// staging. grid=(T/128, B), 8 waves/block, wave owns 16 query rows.
// out = O1/l1 - lam * O2/l2.
// ---------------------------------------------------------------------------
__global__ __launch_bounds__(256) void diff_flash_kernel(
    const unsigned short* __restrict__ qb, const unsigned short* __restrict__ kb,
    const unsigned short* __restrict__ vbt,
    const float* __restrict__ lq1, const float* __restrict__ lq2,
    const float* __restrict__ lk1, const float* __restrict__ lk2,
    float* __restrict__ out) {
  __shared__ __align__(16) unsigned short Ks[2][32 * 128];    // [buf][key][d]
  __shared__ __align__(16) unsigned short Vts[2][128 * 32];   // [buf][col][key]
  __shared__ __align__(16) unsigned short Ps[8][2][16 * 32];

  const int tid = threadIdx.x;
  const int wave = tid >> 5, lane = tid & 31, laneN = lane & 15;
  const int half8 = (lane & 16) ? 8 : 0;
  const int b = blockIdx.y;
  const int qtile = blockIdx.x * 128;
  const int q0w = qtile + wave * 16;

  // lambda = exp(lq1.lk1) - exp(lq2.lk2) + (0.8 - 0.6*exp(-0.3))
  float s1 = 0.f, s2 = 0.f;
#pragma unroll 8
  for (int i = 0; i < HS; ++i) { s1 += lq1[i] * lk1[i]; s2 += lq2[i] * lk2[i]; }
  const float lam = __expf(s1) - __expf(s2) + (0.8f - 0.6f * __expf(-0.3f));

  const unsigned short* qbase   = qb  + (size_t)b * T_SEQ * D2;
  const unsigned short* kbase_g = kb  + (size_t)b * T_SEQ * D2;
  const unsigned short* vbase   = vbt + (size_t)b * D2 * T_SEQ;

  v16bf a1[2], a2[2];
  a1[0] = load_a(qbase, q0w + laneN, D2, 0,  lane);
  a1[1] = load_a(qbase, q0w + laneN, D2, 32, lane);
  a2[0] = load_a(qbase, q0w + laneN, D2, 64, lane);
  a2[1] = load_a(qbase, q0w + laneN, D2, 96, lane);

  v8f vzero = {};
  v8f acc1[8], acc2[8];
#pragma unroll
  for (int o = 0; o < 8; ++o) { acc1[o] = vzero; acc2[o] = vzero; }
  float m1[8], m2[8], l1v[8], l2v[8];
#pragma unroll
  for (int r = 0; r < 8; ++r) { m1[r] = -3.0e38f; m2[r] = -3.0e38f; l1v[r] = 0.f; l2v[r] = 0.f; }

  const float scale = 0.125f;  // 1/sqrt(64)
  const int nblocks = qtile / 32 + 4;   // causal: keys up to qtile+127

  // cooperative stage of one 32-key K / V^T tile into buffer `buf`
  auto stage = [&](int j0, int buf) {
    int r = tid >> 3, c0 = (tid & 7) * 16;
    const unsigned short* ksrc = kbase_g + (size_t)(j0 + r) * D2 + c0;
    unsigned short* kdst = &Ks[buf][r * 128 + c0];
    async_copy16(ksrc, kdst);
    async_copy16(ksrc + 8, kdst + 8);
    int d = tid >> 1, k0 = (tid & 1) * 16;
    const unsigned short* vsrc = vbase + (size_t)d * T_SEQ + j0 + k0;
    unsigned short* vdst = &Vts[buf][d * 32 + k0];
    async_copy16(vsrc, vdst);
    async_copy16(vsrc + 8, vdst + 8);
  };

  stage(0, 0);   // prime buffer 0

  for (int jb = 0; jb < nblocks; ++jb) {
    const int j0 = jb * 32;
    const int cur = jb & 1;
    wait_async_lds();      // this wave's in-flight tile copies done
    __syncthreads();       // everyone's tile is resident; prev readers done
    if (jb + 1 < nblocks) stage((jb + 1) * 32, cur ^ 1);  // overlap next tile

    if (j0 <= q0w + 15) {  // skip fully-masked key blocks for this wave
      const unsigned short* Kb = &Ks[cur][0];
      const unsigned short* Vb = &Vts[cur][0];
      // ---- scores: S = Q K^T over d=64 (2 WMMA steps) per stream/col-tile
      v8f S1[2], S2[2];
#pragma unroll
      for (int c = 0; c < 2; ++c) {
        int key = c * 16 + laneN;
        v16bf bk = load_b(Kb, key, 128, 0, lane);
        v8f s = wmma_bf16(a1[0], bk, vzero);
        bk = load_b(Kb, key, 128, 32, lane);
        S1[c] = wmma_bf16(a1[1], bk, s);
        bk = load_b(Kb, key, 128, 64, lane);
        s = wmma_bf16(a2[0], bk, vzero);
        bk = load_b(Kb, key, 128, 96, lane);
        S2[c] = wmma_bf16(a2[1], bk, s);
      }
      // ---- scale + causal mask (C/D layout: row = r+half8, col = laneN)
#pragma unroll
      for (int c = 0; c < 2; ++c) {
        int krow = j0 + c * 16 + laneN;
#pragma unroll
        for (int r = 0; r < 8; ++r) {
          int qrow = q0w + r + half8;
          float v1 = S1[c][r] * scale, v2 = S2[c][r] * scale;
          if (krow > qrow) { v1 = -3.0e38f; v2 = -3.0e38f; }
          S1[c][r] = v1; S2[c][r] = v2;
        }
      }
      // ---- dual-stream online softmax
      float p1[2][8], p2[2][8];
#pragma unroll
      for (int r = 0; r < 8; ++r) {
        float rm1 = fmaxf(S1[0][r], S1[1][r]);
        float rm2 = fmaxf(S2[0][r], S2[1][r]);
#pragma unroll
        for (int ml = 1; ml <= 8; ml <<= 1) {
          rm1 = fmaxf(rm1, __shfl_xor(rm1, ml, 32));
          rm2 = fmaxf(rm2, __shfl_xor(rm2, ml, 32));
        }
        float mn1 = fmaxf(m1[r], rm1), mn2 = fmaxf(m2[r], rm2);
        float sc1 = __expf(m1[r] - mn1), sc2 = __expf(m2[r] - mn2);
        m1[r] = mn1; m2[r] = mn2;
        float e10 = __expf(S1[0][r] - mn1), e11 = __expf(S1[1][r] - mn1);
        float e20 = __expf(S2[0][r] - mn2), e21 = __expf(S2[1][r] - mn2);
        p1[0][r] = e10; p1[1][r] = e11; p2[0][r] = e20; p2[1][r] = e21;
        float rs1 = e10 + e11, rs2 = e20 + e21;
#pragma unroll
        for (int ml = 1; ml <= 8; ml <<= 1) {
          rs1 += __shfl_xor(rs1, ml, 32);
          rs2 += __shfl_xor(rs2, ml, 32);
        }
        l1v[r] = l1v[r] * sc1 + rs1;
        l2v[r] = l2v[r] * sc2 + rs2;
#pragma unroll
        for (int o = 0; o < 8; ++o) { acc1[o][r] *= sc1; acc2[o][r] *= sc2; }
      }
      // ---- P (C/D layout) -> LDS bf16 -> reload as A fragments
      unsigned short* P1 = &Ps[wave][0][0];
      unsigned short* P2 = &Ps[wave][1][0];
#pragma unroll
      for (int c = 0; c < 2; ++c)
#pragma unroll
        for (int r = 0; r < 8; ++r) {
          int idx = (r + half8) * 32 + c * 16 + laneN;
          P1[idx] = f2bf(p1[c][r]);
          P2[idx] = f2bf(p2[c][r]);
        }
      asm volatile("s_wait_dscnt 0" ::: "memory");  // same-wave LDS RAW
      v16bf ap1 = load_a(P1, laneN, 32, 0, lane);
      v16bf ap2 = load_a(P2, laneN, 32, 0, lane);
      // ---- O += P V (8 col-tiles of 16 over D2=128)
#pragma unroll
      for (int o = 0; o < 8; ++o) {
        v16bf bv = load_b(Vb, o * 16 + laneN, 32, 0, lane);
        acc1[o] = wmma_bf16(ap1, bv, acc1[o]);
        acc2[o] = wmma_bf16(ap2, bv, acc2[o]);
      }
    }
  }

  // epilogue: out = O1/l1 - lam * O2/l2   (f32)
  float* obase = out + (size_t)b * T_SEQ * D2;
#pragma unroll
  for (int o = 0; o < 8; ++o) {
    int col = o * 16 + laneN;
#pragma unroll
    for (int r = 0; r < 8; ++r) {
      int row = q0w + r + half8;
      float o1 = acc1[o][r] / l1v[r];
      float o2 = acc2[o][r] / l2v[r];
      obase[(size_t)row * D2 + col] = o1 - lam * o2;
    }
  }
}

// ---------------------------------------------------------------------------
extern "C" void kernel_launch(void* const* d_in, const int* in_sizes, int n_in,
                              void* d_out, int out_size, void* d_ws, size_t ws_size,
                              hipStream_t stream) {
  const float* x   = (const float*)d_in[0];
  const float* Wq  = (const float*)d_in[1];
  const float* Wk  = (const float*)d_in[2];
  const float* Wv  = (const float*)d_in[3];
  const float* lq1 = (const float*)d_in[4];
  const float* lq2 = (const float*)d_in[5];
  const float* lk1 = (const float*)d_in[6];
  const float* lk2 = (const float*)d_in[7];
  float* out = (float*)d_out;

  // workspace: bf16 q (4MB) | bf16 k (4MB) | bf16 v^T (4MB)
  unsigned short* qb  = (unsigned short*)d_ws;
  unsigned short* kb  = qb + (size_t)4 * T_SEQ * D2;
  unsigned short* vbt = kb + (size_t)4 * T_SEQ * D2;

  qkv_proj_kernel<<<dim3((4 * T_SEQ) / 128, 3), 256, 0, stream>>>(
      x, Wq, Wk, Wv, qb, kb, vbt);
  diff_flash_kernel<<<dim3(T_SEQ / 128, 4), 256, 0, stream>>>(
      qb, kb, vbt, lq1, lq2, lk1, lk2, out);
}
